// BinaryResNetE18_79070347919499
// MI455X (gfx1250) — 8-bit matrix operand order repaired and checked by reading, compiled, not run on hardware
//
#include <hip/hip_runtime.h>
#include <hip/hip_bf16.h>
#include <stdint.h>

typedef int v8i __attribute__((ext_vector_type(8)));
typedef int v4i __attribute__((ext_vector_type(4)));
typedef int v2i __attribute__((ext_vector_type(2)));

#define BN_EPS 1e-5f

__global__ void stem_conv7x7(const float* __restrict__ x, const float* __restrict__ w,
                             float* __restrict__ out) {
    const int CO = 64, WO = 112, HO = 112, WI = 224, HI = 224, CI = 3;
    long idx = (long)blockIdx.x * blockDim.x + threadIdx.x;
    long total = 32L * HO * WO * CO;
    if (idx >= total) return;
    int co = idx % CO; long t = idx / CO;
    int ox = t % WO; t /= WO; int oy = t % HO; int b = (int)(t / HO);
    float acc = 0.f;
    for (int ky = 0; ky < 7; ++ky) {
        int iy = oy * 2 + ky - 2;
        if ((unsigned)iy >= (unsigned)HI) continue;
        for (int kx = 0; kx < 7; ++kx) {
            int ix = ox * 2 + kx - 2;
            if ((unsigned)ix >= (unsigned)WI) continue;
            const float* xp = x + (((long)b * HI + iy) * WI + ix) * CI;
            const float* wp = w + ((ky * 7 + kx) * CI) * CO + co;
            acc += xp[0] * wp[0] + xp[1] * wp[CO] + xp[2] * wp[2 * CO];
        }
    }
    out[idx] = acc;
}

__global__ void bn_act(const float* __restrict__ in, float* __restrict__ out,
                       const float* __restrict__ gamma, const float* __restrict__ beta,
                       const float* __restrict__ mean, const float* __restrict__ var,
                       int C, long total, int relu) {
    long i = (long)blockIdx.x * blockDim.x + threadIdx.x;
    if (i >= total) return;
    int c = (int)(i % C);
    float s = gamma[c] * rsqrtf(var[c] + BN_EPS);
    float v = (in[i] - mean[c]) * s + beta[c];
    if (relu) v = fmaxf(v, 0.f);
    out[i] = v;
}

__global__ void maxpool3s2(const float* __restrict__ in, float* __restrict__ out,
                           int B, int HI, int WI, int C) {
    int HO = HI / 2, WO = WI / 2;
    long total = (long)B * HO * WO * C;
    long idx = (long)blockIdx.x * blockDim.x + threadIdx.x;
    if (idx >= total) return;
    int c = (int)(idx % C); long t = idx / C;
    int ox = t % WO; t /= WO; int oy = t % HO; int b = (int)(t / HO);
    float m = -INFINITY;
    for (int ky = 0; ky < 3; ++ky) {
        int iy = 2 * oy + ky; if (iy >= HI) continue;
        for (int kx = 0; kx < 3; ++kx) {
            int ix = 2 * ox + kx; if (ix >= WI) continue;
            m = fmaxf(m, in[(((long)b * HI + iy) * WI + ix) * C + c]);
        }
    }
    out[idx] = m;
}

__global__ void avgpool2k(const float* __restrict__ in, float* __restrict__ out,
                          int B, int HI, int WI, int C) {
    int HO = HI / 2, WO = WI / 2;
    long total = (long)B * HO * WO * C;
    long idx = (long)blockIdx.x * blockDim.x + threadIdx.x;
    if (idx >= total) return;
    int c = (int)(idx % C); long t = idx / C;
    int ox = t % WO; t /= WO; int oy = t % HO; int b = (int)(t / HO);
    const float* p = in + (((long)b * HI + 2 * oy) * WI + 2 * ox) * C + c;
    float s = p[0] + p[C] + p[(long)WI * C] + p[(long)WI * C + C];
    out[idx] = s * 0.25f;
}

__global__ void conv1x1_bn(const float* __restrict__ in, const float* __restrict__ w,
                           float* __restrict__ out,
                           const float* __restrict__ gamma, const float* __restrict__ beta,
                           const float* __restrict__ mean, const float* __restrict__ var,
                           int M, int Cin, int Cout) {
    long total = (long)M * Cout;
    long idx = (long)blockIdx.x * blockDim.x + threadIdx.x;
    if (idx >= total) return;
    int co = (int)(idx % Cout); long m = idx / Cout;
    float acc = 0.f;
    const float* ip = in + m * Cin;
    for (int ci = 0; ci < Cin; ++ci) acc += ip[ci] * w[(long)ci * Cout + co];
    float s = gamma[co] * rsqrtf(var[co] + BN_EPS);
    out[idx] = (acc - mean[co]) * s + beta[co];
}

__global__ void fill_zero(int* __restrict__ p, long n) {
    long i = (long)blockIdx.x * blockDim.x + threadIdx.x;
    if (i < n) p[i] = 0;
}

__global__ void binarize_pad(const float* __restrict__ h, int8_t* __restrict__ apad,
                             int B, int H, int W, int C) {
    int Hp = H + 2, Wp = W + 2;
    long total = (long)B * Hp * Wp * C;
    long idx = (long)blockIdx.x * blockDim.x + threadIdx.x;
    if (idx >= total) return;
    int c = (int)(idx % C); long t = idx / C;
    int xp = t % Wp; t /= Wp; int yp = t % Hp; int b = (int)(t / Hp);
    int8_t v = 0;
    if (yp >= 1 && yp <= H && xp >= 1 && xp <= W) {
        float f = h[(((long)b * H + (yp - 1)) * W + (xp - 1)) * C + c];
        v = (f >= 0.f) ? (int8_t)1 : (int8_t)-1;
    }
    apad[idx] = v;
}

__global__ void pack_weights(const float* __restrict__ w, int* __restrict__ wpack,
                             int Cin, int Cout) {
    int KC = Cin / 64, NT = Cout / 16;
    long total = 9L * KC * NT * 256;
    long idx = (long)blockIdx.x * blockDim.x + threadIdx.x;
    if (idx >= total) return;
    int v = (int)(idx & 7);
    int lane = (int)((idx >> 3) & 31);
    long r = idx >> 8;
    int nt = (int)(r % NT); r /= NT;
    int kc = (int)(r % KC); int tap = (int)(r / KC);
    int half = lane >> 4, col = lane & 15;
    int kb = (v & 1) * 4 + half * 8 + (v >> 1) * 16;
    int co = nt * 16 + col;
    unsigned d = 0;
    for (int j = 0; j < 4; ++j) {
        int ci = kc * 64 + kb + j;
        float f = w[((long)tap * Cin + ci) * Cout + co];
        unsigned char s = (f >= 0.f) ? (unsigned char)1 : (unsigned char)0xFF;
        d |= ((unsigned)s) << (8 * j);
    }
    wpack[idx] = (int)d;
}

template <int CIN, int COUT, int STRIDE, int H>
__global__ void __launch_bounds__(128)
bconv_wmma(const int8_t* __restrict__ apad, const int* __restrict__ wpack,
           const float* __restrict__ resid, float* __restrict__ out,
           int8_t* __restrict__ spad_next,
           const float* __restrict__ gamma, const float* __restrict__ beta,
           const float* __restrict__ mean, const float* __restrict__ var) {
    constexpr int KC = CIN / 64, NT = COUT / 16;
    constexpr int HO = H / STRIDE, WO = HO;
    constexpr int HP = H + 2, WP = H + 2;
    constexpr int MT = (32 * HO * WO) / 16;
    constexpr int PADB = STRIDE - 1;

    const int lane = threadIdx.x;
    const int mtile = blockIdx.x * 4 + threadIdx.y;
    if (mtile >= MT) return;
    const int ngrp = blockIdx.y;

    const int mrow = mtile * 16 + (lane & 15);
    const int ox = mrow % WO;
    const int oy = (mrow / WO) % HO;
    const int b  = mrow / (WO * HO);
    const int half = lane >> 4;

    v8i zero = {0, 0, 0, 0, 0, 0, 0, 0};
    v8i acc[4];
#pragma unroll
    for (int j = 0; j < 4; ++j) acc[j] = zero;

#pragma unroll
    for (int ky = 0; ky < 3; ++ky) {
        const int iyp = oy * STRIDE + ky + PADB;
#pragma unroll
        for (int kx = 0; kx < 3; ++kx) {
            const int ixp = ox * STRIDE + kx + PADB;
            const int8_t* arow =
                apad + (((long)b * HP + iyp) * WP + ixp) * (long)CIN + half * 8;
            const int tap = ky * 3 + kx;
            for (int kc = 0; kc < KC; ++kc) {
                const int8_t* ap = arow + kc * 64;
                v2i a01 = *reinterpret_cast<const v2i*>(ap);
                v2i a23 = *reinterpret_cast<const v2i*>(ap + 16);
                v2i a45 = *reinterpret_cast<const v2i*>(ap + 32);
                v2i a67 = *reinterpret_cast<const v2i*>(ap + 48);
                v8i A;
                A[0] = a01[0]; A[1] = a01[1];
                A[2] = a23[0]; A[3] = a23[1];
                A[4] = a45[0]; A[5] = a45[1];
                A[6] = a67[0]; A[7] = a67[1];

                const int* bp0 =
                    wpack + ((((long)tap * KC + kc) * NT + ngrp * 4) * 32 + lane) * 8;
                __builtin_prefetch(bp0 + 1024, 0, 1);
#pragma unroll
                for (int j = 0; j < 4; ++j) {
                    const int* bp = bp0 + j * 256;
                    v4i b0 = *reinterpret_cast<const v4i*>(bp);
                    v4i b1 = *reinterpret_cast<const v4i*>(bp + 4);
                    v8i Bf;
                    Bf[0] = b0[0]; Bf[1] = b0[1]; Bf[2] = b0[2]; Bf[3] = b0[3];
                    Bf[4] = b1[0]; Bf[5] = b1[1]; Bf[6] = b1[2]; Bf[7] = b1[3];
                    acc[j] = __builtin_amdgcn_wmma_i32_16x16x64_iu8(
                        true, A, true, Bf, acc[j], false, false);
                }
            }
        }
    }

    float sj[4], bj[4];
    int coj[4];
#pragma unroll
    for (int j = 0; j < 4; ++j) {
        int co = (ngrp * 4 + j) * 16 + (lane & 15);
        coj[j] = co;
        float s = gamma[co] * rsqrtf(var[co] + BN_EPS);
        sj[j] = s;
        bj[j] = beta[co] - mean[co] * s;
    }
#pragma unroll
    for (int r = 0; r < 8; ++r) {
        const int m = mtile * 16 + r + 8 * half;
        const int mx = m % WO;
        const int my = (m / WO) % HO;
        const int mb = m / (WO * HO);
        const long prow =
            (((long)mb * (HO + 2) + (my + 1)) * (WO + 2) + (mx + 1)) * (long)COUT;
#pragma unroll
        for (int j = 0; j < 4; ++j) {
            const long o = (long)m * COUT + coj[j];
            const float val = (float)acc[j][r] * sj[j] + bj[j] + resid[o];
            out[o] = val;
            if (spad_next)
                spad_next[prow + coj[j]] = (val >= 0.f) ? (int8_t)1 : (int8_t)-1;
        }
    }
}

__global__ void relu_gap(const float* __restrict__ in, float* __restrict__ out,
                         int B, int HW, int C) {
    long total = (long)B * C;
    long idx = (long)blockIdx.x * blockDim.x + threadIdx.x;
    if (idx >= total) return;
    int c = (int)(idx % C); int b = (int)(idx / C);
    float s = 0.f;
    for (int i = 0; i < HW; ++i) s += fmaxf(in[((long)b * HW + i) * C + c], 0.f);
    out[idx] = s / (float)HW;
}

__global__ void head_gemm(const float* __restrict__ h, const float* __restrict__ w,
                          const float* __restrict__ bias, float* __restrict__ out,
                          int B, int C, int N) {
    long total = (long)B * N;
    long idx = (long)blockIdx.x * blockDim.x + threadIdx.x;
    if (idx >= total) return;
    int n = (int)(idx % N); int b = (int)(idx / N);
    float acc = bias[n];
    for (int c = 0; c < C; ++c) acc += h[(long)b * C + c] * w[(long)c * N + n];
    out[idx] = acc;
}

__global__ void softmax_rows(const float* __restrict__ in, float* __restrict__ out, int N) {
    __shared__ float red[256];
    int tid = threadIdx.x;
    const float* row = in + (long)blockIdx.x * N;
    float m = -INFINITY;
    for (int i = tid; i < N; i += 256) m = fmaxf(m, row[i]);
    red[tid] = m; __syncthreads();
    for (int s = 128; s > 0; s >>= 1) {
        if (tid < s) red[tid] = fmaxf(red[tid], red[tid + s]);
        __syncthreads();
    }
    m = red[0]; __syncthreads();
    float sum = 0.f;
    for (int i = tid; i < N; i += 256) sum += expf(row[i] - m);
    red[tid] = sum; __syncthreads();
    for (int s = 128; s > 0; s >>= 1) {
        if (tid < s) red[tid] += red[tid + s];
        __syncthreads();
    }
    float inv = 1.f / red[0];
    for (int i = tid; i < N; i += 256)
        out[(long)blockIdx.x * N + i] = expf(row[i] - m) * inv;
}

template <int CIN, int COUT, int STRIDE, int H>
static void launch_bconv(const int8_t* apad, const int* wpack, const float* resid,
                         float* out, int8_t* spadn,
                         const float* g, const float* be, const float* me, const float* va,
                         hipStream_t stream) {
    constexpr int HO = H / STRIDE;
    constexpr int MT = (32 * HO * HO) / 16;
    constexpr int NT = COUT / 16;
    dim3 grid((MT + 3) / 4, NT / 4), blk(32, 4);
    bconv_wmma<CIN, COUT, STRIDE, H>
        <<<grid, blk, 0, stream>>>(apad, wpack, resid, out, spadn, g, be, me, va);
}

extern "C" void kernel_launch(void* const* d_in, const int* in_sizes, int n_in,
                              void* d_out, int out_size, void* d_ws, size_t ws_size,
                              hipStream_t stream) {
    (void)in_sizes; (void)n_in; (void)out_size; (void)ws_size;

    int bn_i[16][4], dbn_i[16][4], dw_i[16], w_i[16];
    bool down[16];
    for (int i = 0; i < 16; ++i) down[i] = (i == 4 || i == 8 || i == 12);
    int cur = 0;
    for (int i = 0; i < 16; ++i) {
        for (int j = 0; j < 4; ++j) bn_i[i][j] = cur++;
        if (down[i]) {
            for (int j = 0; j < 4; ++j) dbn_i[i][j] = cur++;
            dw_i[i] = cur++;
        } else { dw_i[i] = -1; }
        w_i[i] = cur++;
    }
    int head_b_i = cur++, head_w_i = cur++;
    int sbn1_i = cur; cur += 4;
    int sbn2_i = cur; cur += 4;
    int stem_w_i = cur++;
    int x_i = cur++;

    auto F = [&](int i) { return (const float*)d_in[i]; };

    char* ws = (char*)d_ws;
    float*  stem   = (float*)(ws + 0ull);
    float*  hA     = (float*)(ws + 104857600ull);
    float*  hB     = (float*)(ws + 138412032ull);
    float*  res    = (float*)(ws + 171966464ull);
    int8_t* apadA  = (int8_t*)(ws + 205520896ull);
    int8_t* apadB  = (int8_t*)(ws + 213909504ull);
    int*    wpack  = (int*)  (ws + 222298112ull);
    float*  pool   = (float*)(ws + 226492416ull);
    float*  gap    = (float*)(ws + 234881024ull);
    float*  logits = (float*)(ws + 235143168ull);

    const int B = 32;
    auto cdiv = [](long a, long b) { return (unsigned)((a + b - 1) / b); };

    {
        long tot = (long)B * 112 * 112 * 64;
        stem_conv7x7<<<cdiv(tot, 256), 256, 0, stream>>>(F(x_i), F(stem_w_i), stem);
        bn_act<<<cdiv(tot, 256), 256, 0, stream>>>(stem, stem,
            F(sbn1_i + 1), F(sbn1_i + 0), F(sbn1_i + 2), F(sbn1_i + 3), 64, tot, 1);
        long tot2 = (long)B * 56 * 56 * 64;
        maxpool3s2<<<cdiv(tot2, 256), 256, 0, stream>>>(stem, hA, B, 112, 112, 64);
        bn_act<<<cdiv(tot2, 256), 256, 0, stream>>>(hA, hA,
            F(sbn2_i + 1), F(sbn2_i + 0), F(sbn2_i + 2), F(sbn2_i + 3), 64, tot2, 0);
    }

    int Hs[16], Cin_[16], Cout_[16], st_[16];
    {
        int cin = 64, H = 56, bi = 0;
        int stage_f[4] = {64, 128, 256, 512};
        for (int s = 0; s < 4; ++s)
            for (int l = 0; l < 4; ++l) {
                int f = stage_f[s];
                Hs[bi] = H; Cin_[bi] = cin; Cout_[bi] = f;
                st_[bi] = (cin != f) ? 2 : 1;
                if (cin != f) H /= 2;
                cin = f; ++bi;
            }
    }

    {
        long atot = (long)B * 58 * 58 * 64;
        binarize_pad<<<cdiv(atot, 256), 256, 0, stream>>>(hA, apadA, B, 56, 56, 64);
    }

    float* hcur = hA; float* hnext = hB;
    int8_t* acur = apadA; int8_t* anxt = apadB;
    for (int i = 0; i < 16; ++i) {
        int H = Hs[i], Cin = Cin_[i], Cout = Cout_[i], stride = st_[i];
        int Ho = H / stride;
        long M = (long)B * Ho * Ho;
        const float* residPtr;
        if (down[i]) {
            long ptot = (long)B * Ho * Ho * Cin;
            avgpool2k<<<cdiv(ptot, 256), 256, 0, stream>>>(hcur, pool, B, H, H, Cin);
            long dtot = M * Cout;
            conv1x1_bn<<<cdiv(dtot, 256), 256, 0, stream>>>(pool, F(dw_i[i]), res,
                F(dbn_i[i][1]), F(dbn_i[i][0]), F(dbn_i[i][2]), F(dbn_i[i][3]),
                (int)M, Cin, Cout);
            residPtr = res;
        } else {
            residPtr = hcur;
        }
        int KC = Cin / 64, NT = Cout / 16;
        long wtot = 9L * KC * NT * 256;
        pack_weights<<<cdiv(wtot, 256), 256, 0, stream>>>(F(w_i[i]), wpack, Cin, Cout);

        int8_t* spadn = nullptr;
        if (i < 15) {
            long ndw = ((long)B * (Ho + 2) * (Ho + 2) * Cout) / 4;
            fill_zero<<<cdiv(ndw, 256), 256, 0, stream>>>((int*)anxt, ndw);
            spadn = anxt;
        }

        const float* g  = F(bn_i[i][1]);
        const float* be = F(bn_i[i][0]);
        const float* me = F(bn_i[i][2]);
        const float* va = F(bn_i[i][3]);
        if (i < 4)        launch_bconv< 64,  64, 1, 56>(acur, wpack, residPtr, hnext, spadn, g, be, me, va, stream);
        else if (i == 4)  launch_bconv< 64, 128, 2, 56>(acur, wpack, residPtr, hnext, spadn, g, be, me, va, stream);
        else if (i < 8)   launch_bconv<128, 128, 1, 28>(acur, wpack, residPtr, hnext, spadn, g, be, me, va, stream);
        else if (i == 8)  launch_bconv<128, 256, 2, 28>(acur, wpack, residPtr, hnext, spadn, g, be, me, va, stream);
        else if (i < 12)  launch_bconv<256, 256, 1, 14>(acur, wpack, residPtr, hnext, spadn, g, be, me, va, stream);
        else if (i == 12) launch_bconv<256, 512, 2, 14>(acur, wpack, residPtr, hnext, spadn, g, be, me, va, stream);
        else              launch_bconv<512, 512, 1,  7>(acur, wpack, residPtr, hnext, spadn, g, be, me, va, stream);

        float* t = hcur; hcur = hnext; hnext = t;
        int8_t* u = acur; acur = anxt; anxt = u;
    }

    relu_gap<<<cdiv((long)B * 512, 256), 256, 0, stream>>>(hcur, gap, B, 49, 512);
    head_gemm<<<cdiv((long)B * 1000, 256), 256, 0, stream>>>(gap, F(head_w_i), F(head_b_i),
                                                             logits, B, 512, 1000);
    softmax_rows<<<B, 256, 0, stream>>>(logits, (float*)d_out, 1000);
}
